// MultiHeadAttention_7421703487711
// MI455X (gfx1250) — matrix operand order repaired and checked by reading, not compiled, not run on hardware
//
#include <hip/hip_runtime.h>
#include <hip/hip_bf16.h>

// ---------------------------------------------------------------------------
// MHA block for MI455X (gfx1250, wave32, WMMA + Tensor Data Mover).
// All matmuls: v_wmma_f32_16x16x32_bf16 (f32 accumulate).
// GEMM weight tiles are staged to LDS by the TDM (tensor_load_to_lds,
// TENSORcnt double-buffered); activations are pre-converted to bf16 so the
// staging is a pure byte copy. The 1 GB attn materialization uses
// non-temporal stores (streaming, larger than the 192 MB L2).
//
// Workspace layout (assumes ws_size >= 106 MiB), with phase-based reuse:
//   [  0 MiB) xq  bf16 [M,1024]  -> reused later: xbuf f32 [M,1024] (32 MiB)
//   [ 16 MiB) xk  bf16 [M,1024]
//   [ 32 MiB) xv  bf16 [M,1024]  -> reused later: att bf16 [M,1024]
//   [ 48 MiB) wqb/wkb/wvb/wob bf16 [1024,1024] (2 MiB each)
//   [ 56 MiB) qh  bf16 [B,H,L,64]
//   [ 72 MiB) kh  bf16 [B,H,L,64]
//   [ 88 MiB) vht bf16 [B,H,64,L]
//   [104 MiB) mbits u32 [B*L, L/32] packed (edge==0||mask) bitmask (2 MiB)
// ---------------------------------------------------------------------------

typedef __bf16 bf16_t;
typedef __attribute__((ext_vector_type(16))) __bf16 v16bf;
typedef __attribute__((ext_vector_type(4)))  __bf16 v4bf;
typedef __attribute__((ext_vector_type(8)))  float  v8f;
typedef __attribute__((ext_vector_type(4)))  unsigned u32x4;
typedef __attribute__((ext_vector_type(8)))  unsigned u32x8;

constexpr int Bc = 4, Lc = 2048, Dc = 1024, Hc = 16;
constexpr int Mc = Bc * Lc;  // 8192 rows
constexpr float NEGV = -1e15f;

// --- WMMA fragment helpers (A and B both use contiguous k = g*16 + s) -------

// A fragment (16x32 bf16, MxK): lane holds row m=lane&15; VGPR v holds the
// K-pair at k0 = g*16 + v*2  (g = lane>>4). Pairs = 4B loads.
__device__ inline v16bf load_a_frag(const bf16_t* __restrict__ p, int g) {
  v16bf f;
#pragma unroll
  for (int v = 0; v < 8; ++v) {
    const int k0 = g * 16 + v * 2;
    f[2 * v]     = p[k0];
    f[2 * v + 1] = p[k0 + 1];
  }
  return f;
}

// B fragment (32x16 bf16, KxN): lane holds col n=lane&15; K values are
// g*16 + 0..15, i.e. 16 contiguous bf16 (two b128 loads) when the tile is
// K-contiguous per column. p points at the (col, kchunk) base.
__device__ inline v16bf load_b_frag(const bf16_t* __restrict__ p, int g) {
  return *reinterpret_cast<const v16bf*>(p + g * 16);
}

__device__ inline v8f wmma_bf16(v16bf a, v16bf b, v8f c) {
  return __builtin_amdgcn_wmma_f32_16x16x32_bf16(
      /*neg_a=*/false, a, /*neg_b=*/false, b,
      /*c_mod=*/(short)0, c, /*reuse_a=*/false, /*reuse_b=*/false);
}

// --- Tensor Data Mover: 2D bf16 tile (rows x cols) global -> LDS ------------
// D# groups 0/1 built per CDNA5 ISA:
//  g0: count=1 | lds_addr | global_addr[56:0] | type=2
//  g1: data_size=1(2B) | tensor_dim0/1 | tile_dim0/1 | tensor_dim0_stride
// Issued as a wave-scalar op (gate with wave id); tracked by TENSORcnt.
__device__ inline void tdm_load_tile_bf16(unsigned lds_off,
                                          const bf16_t* __restrict__ gsrc,
                                          unsigned rows, unsigned cols,
                                          unsigned row_stride_elems) {
  const unsigned long long ga = (unsigned long long)gsrc;
  u32x4 g0;
  g0[0] = 1u;                                           // count=1, user mode
  g0[1] = lds_off;                                      // lds_addr (bytes)
  g0[2] = (unsigned)(ga & 0xffffffffu);                 // global_addr[31:0]
  g0[3] = ((unsigned)(ga >> 32) & 0x01ffffffu)          // global_addr[56:32]
          | 0x80000000u;                                // type=2 ("image")
  u32x8 g1;
  g1[0] = 1u << 16;            // workgroup_mask=0 | data_size=1 (2 bytes)
  g1[1] = cols << 16;          // tensor_dim0[15:0]  (bits [79:48])
  g1[2] = rows << 16;          // tensor_dim0 hi=0 | tensor_dim1 lo
  g1[3] = cols << 16;          // tensor_dim1 hi=0 | tile_dim0
  g1[4] = rows;                // tile_dim1 | tile_dim2=0
  g1[5] = row_stride_elems;    // tensor_dim0_stride[31:0] (data_size units)
  g1[6] = 0u;                  // stride hi | tensor_dim1_stride lo
  g1[7] = 0u;
  asm volatile("tensor_load_to_lds %0, %1" ::"s"(g0), "s"(g1) : "memory");
}

// ---------------------------------------------------------------------------
// Kernel 0: f32 -> bf16 bulk convert (float4 in, 4x bf16 out).
// ---------------------------------------------------------------------------
__global__ __launch_bounds__(256) void cvt_bf16_kernel(
    const float* __restrict__ s, bf16_t* __restrict__ d) {
  const size_t i = ((size_t)blockIdx.x * 256 + threadIdx.x) * 4;
  const float4 v = *reinterpret_cast<const float4*>(s + i);
  v4bf o;
  o[0] = (bf16_t)v.x; o[1] = (bf16_t)v.y;
  o[2] = (bf16_t)v.z; o[3] = (bf16_t)v.w;
  *reinterpret_cast<v4bf*>(d + i) = o;
}

// ---------------------------------------------------------------------------
// Kernel 1: GEMM  acc[M,N] = A[M,K=1024] * W[N,K]^T  (bf16 in, f32 acc).
// Block: 256 threads / 8 waves, tile 128(M) x 64(N), K-step 32.
// W tile (64x32 bf16 = 4 KB) is TDM-loaded into LDS, double-buffered on
// TENSORcnt. A-fragments load straight from global bf16 (L2-resident reuse).
// Epilogues: 0 = bf16 head-split [B,H,L,64]; 1 = bf16 transposed [B,H,64,L];
//            2 = f32 + bias + residual (pre-LayerNorm).
// ---------------------------------------------------------------------------
template <int MODE>
__global__ __launch_bounds__(256) void gemm_kernel(
    const bf16_t* __restrict__ A, const bf16_t* __restrict__ W,
    const float* __restrict__ bias, const float* __restrict__ resid,
    bf16_t* __restrict__ outb, float* __restrict__ outf) {
  __shared__ bf16_t sB[2][64 * 32];
  const int tid = threadIdx.x, lane = tid & 31, wave = tid >> 5;
  const int g = lane >> 4, nl = lane & 15;
  const int m0 = blockIdx.y * 128, n0 = blockIdx.x * 64;
  const bf16_t* arow = A + (size_t)(m0 + wave * 16 + nl) * Dc;
  const bf16_t* wbase = W + (size_t)n0 * Dc;
  constexpr int KT = Dc / 32;

  if (wave == 0)
    tdm_load_tile_bf16((unsigned)(unsigned long long)&sB[0][0], wbase, 64, 32,
                       Dc);

  v8f acc[4] = {};
  for (int it = 0; it < KT; ++it) {
    const int kk = it * 32, cur = it & 1;
    if (wave == 0) {
      if (it + 1 < KT) {
        tdm_load_tile_bf16((unsigned)(unsigned long long)&sB[cur ^ 1][0],
                           wbase + kk + 32, 64, 32, Dc);
        __builtin_amdgcn_s_wait_tensorcnt((short)1);  // previous tile landed
      } else {
        __builtin_amdgcn_s_wait_tensorcnt((short)0);
      }
    }
    __syncthreads();
    if (it + 4 < KT) __builtin_prefetch(arow + kk + 128, 0, 0);
    const v16bf a = load_a_frag(arow + kk, g);
#pragma unroll
    for (int nt = 0; nt < 4; ++nt) {
      const v16bf bfr = load_b_frag(&sB[cur][(nt * 16 + nl) * 32], g);
      acc[nt] = wmma_bf16(a, bfr, acc[nt]);
    }
    __syncthreads();
  }

  // C/D layout: n = lane&15, m = r + g*8.
#pragma unroll
  for (int nt = 0; nt < 4; ++nt) {
#pragma unroll
    for (int r = 0; r < 8; ++r) {
      const int gm = m0 + wave * 16 + r + g * 8;
      const int gn = n0 + nt * 16 + nl;
      const float v = acc[nt][r] + bias[gn];
      if (MODE == 2) {
        outf[(size_t)gm * Dc + gn] = v + resid[(size_t)gm * Dc + gn];
      } else {
        const int b = gm >> 11, l = gm & (Lc - 1);
        const int h = gn >> 6, d = gn & 63;
        if (MODE == 1)
          outb[(((size_t)(b * Hc + h)) * 64 + d) * Lc + l] = (bf16_t)v;
        else
          outb[(((size_t)(b * Hc + h)) * Lc + l) * 64 + d] = (bf16_t)v;
      }
    }
  }
}

// ---------------------------------------------------------------------------
// Kernel 2: pack (edge==0 || mask) into 1 bit per (b,q,k); masks are reused
// by all 16 heads, so this cuts attention-phase mask traffic 40x.
// ---------------------------------------------------------------------------
__global__ __launch_bounds__(256) void maskpack_kernel(
    const int* __restrict__ edge, const unsigned char* __restrict__ mask,
    unsigned int* __restrict__ mbits) {
  const size_t w = (size_t)blockIdx.x * 256 + threadIdx.x;
  const size_t base = w * 32;
  unsigned int bits = 0u;
#pragma unroll
  for (int i = 0; i < 32; ++i)
    if (edge[base + i] == 0 || mask[base + i]) bits |= (1u << i);
  mbits[w] = bits;
}

// ---------------------------------------------------------------------------
// Kernel 3: fused attention per (b, h, 16-query tile). 256 threads / 8 waves.
// Phase 1: S = Q K^T / 8 with mask -> 16x2048 f32 strip in LDS (128 KB).
// Phase 2: row softmax in LDS; attn -> d_out (non-temporal); P -> bf16 LDS.
// Phase 3: O = P V via WMMA (split-K over wave pairs, LDS reduce).
// Dynamic LDS: 128 KB (S, reused for split-K partials) + 64 KB (P) = 192 KB
// of the 320 KB/WGP budget.
// ---------------------------------------------------------------------------
__global__ __launch_bounds__(256) void attn_kernel(
    const bf16_t* __restrict__ qh, const bf16_t* __restrict__ kh,
    const bf16_t* __restrict__ vht, const unsigned int* __restrict__ mbits,
    float* __restrict__ attn_out, bf16_t* __restrict__ att_ws) {
  extern __shared__ char smem[];
  float*  sS = (float*)smem;                   // [16][2048] f32
  bf16_t* sP = (bf16_t*)(smem + 16 * Lc * 4);  // [16][2048] bf16
  float*  sRed = sS;                           // reused: [8][32][8] f32

  const int tid = threadIdx.x, lane = tid & 31, wave = tid >> 5;
  const int g = lane >> 4, nl = lane & 15;
  const int qt = blockIdx.x & (Lc / 16 - 1);
  const int bh = blockIdx.x >> 7;  // b*H + h
  const int bat = bh >> 4, h = bh & (Hc - 1);
  const int q0 = qt * 16;
  const float inv_t = 0.125f;  // 1/sqrt(DK)

  // ---- Phase 1: scores ----
  const bf16_t* qbase = qh + ((size_t)bh * Lc + q0) * 64 + nl * 64;
  const v16bf aq0 = load_a_frag(qbase, g);       // d = 0..31
  const v16bf aq1 = load_a_frag(qbase + 32, g);  // d = 32..63
  const bf16_t* kbase = kh + (size_t)bh * Lc * 64;
  for (int kt = wave; kt < Lc / 16; kt += 8) {
    const bf16_t* kb = kbase + (size_t)(kt * 16 + nl) * 64;
    v8f s = {};
    s = wmma_bf16(aq0, load_b_frag(kb, g), s);
    s = wmma_bf16(aq1, load_b_frag(kb + 32, g), s);
    const int key = kt * 16 + nl;
#pragma unroll
    for (int r = 0; r < 8; ++r) {
      const int m = r + g * 8, q = q0 + m;
      const unsigned int w =
          mbits[(size_t)(bat * Lc + q) * (Lc / 32) + (key >> 5)];
      float v = s[r] * inv_t;
      if ((w >> (key & 31)) & 1u) v = NEGV;
      sS[m * Lc + key] = v;
    }
  }
  __syncthreads();

  // ---- Phase 2: softmax (2 rows per wave, lane-strided over 2048 keys) ----
  for (int rr = 0; rr < 2; ++rr) {
    const int m = wave * 2 + rr, q = q0 + m;
    float* srow = sS + m * Lc;
    float mx = -3.4e38f;
    for (int k = lane; k < Lc; k += 32) mx = fmaxf(mx, srow[k]);
#pragma unroll
    for (int o = 16; o > 0; o >>= 1) mx = fmaxf(mx, __shfl_xor(mx, o, 32));
    float sum = 0.f;
    for (int k = lane; k < Lc; k += 32) {
      const float p = __expf(srow[k] - mx);
      srow[k] = p;
      sum += p;
    }
#pragma unroll
    for (int o = 16; o > 0; o >>= 1) sum += __shfl_xor(sum, o, 32);
    const float inv = 1.f / sum;
    float* arow = attn_out + ((size_t)(h * Bc + bat) * Lc + q) * Lc;
    bf16_t* prow = sP + m * Lc;
    for (int k = lane; k < Lc; k += 32) {
      const float p = srow[k] * inv;
      __builtin_nontemporal_store(p, arow + k);  // streaming 1 GB output
      prow[k] = (bf16_t)p;
    }
  }
  __syncthreads();

  // ---- Phase 3: O = P V. wave -> (ntile = w&3, khalf = w>>2) split-K ----
  const int ntile = wave & 3, khalf = wave >> 2;
  const bf16_t* vb = vht + ((size_t)bh * 64 + ntile * 16 + nl) * Lc;
  v8f acc = {};
  for (int ks = khalf * 32; ks < khalf * 32 + 32; ++ks) {
    const v16bf a = load_a_frag(sP + nl * Lc + ks * 32, g);
    const v16bf bfr = load_b_frag(vb + ks * 32, g);
    acc = wmma_bf16(a, bfr, acc);
  }
#pragma unroll
  for (int r = 0; r < 8; ++r) sRed[(wave * 32 + lane) * 8 + r] = acc[r];
  __syncthreads();
  if (wave < 4) {
#pragma unroll
    for (int r = 0; r < 8; ++r)
      acc[r] += sRed[((wave + 4) * 32 + lane) * 8 + r];
#pragma unroll
    for (int r = 0; r < 8; ++r) {
      const int m = r + g * 8;
      const size_t row = (size_t)bat * Lc + q0 + m;
      const int col = h * 64 + ntile * 16 + nl;
      att_ws[row * (Hc * 64) + col] = (bf16_t)acc[r];
    }
  }
}

// ---------------------------------------------------------------------------
// Kernel 4: LayerNorm over D=1024 per row. One 256-thread block per row.
// ---------------------------------------------------------------------------
__global__ __launch_bounds__(256) void ln_kernel(
    const float* __restrict__ x, const float* __restrict__ lng,
    const float* __restrict__ lnb, float* __restrict__ y) {
  __shared__ float red[16];
  const int row = blockIdx.x;
  const int lane = threadIdx.x & 31, wave = threadIdx.x >> 5;
  const float* xr = x + (size_t)row * Dc;
  float v[4], s = 0.f, s2 = 0.f;
#pragma unroll
  for (int i = 0; i < 4; ++i) {
    v[i] = xr[threadIdx.x + i * 256];
    s += v[i];
    s2 += v[i] * v[i];
  }
#pragma unroll
  for (int o = 16; o > 0; o >>= 1) {
    s += __shfl_xor(s, o, 32);
    s2 += __shfl_xor(s2, o, 32);
  }
  if (lane == 0) { red[wave] = s; red[8 + wave] = s2; }
  __syncthreads();
  if (wave == 0) {
    float a = (lane < 8) ? red[lane] : 0.f;
    float b = (lane < 8) ? red[8 + lane] : 0.f;
#pragma unroll
    for (int o = 4; o > 0; o >>= 1) {
      a += __shfl_xor(a, o, 32);
      b += __shfl_xor(b, o, 32);
    }
    if (lane == 0) { red[0] = a; red[1] = b; }
  }
  __syncthreads();
  const float mu = red[0] * (1.f / Dc);
  const float var = red[1] * (1.f / Dc) - mu * mu;
  const float rinv = rsqrtf(var + 1e-5f);
#pragma unroll
  for (int i = 0; i < 4; ++i) {
    const int c = threadIdx.x + i * 256;
    y[(size_t)row * Dc + c] = (v[i] - mu) * rinv * lng[c] + lnb[c];
  }
}

// ---------------------------------------------------------------------------
extern "C" void kernel_launch(void* const* d_in, const int* in_sizes, int n_in,
                              void* d_out, int out_size, void* d_ws,
                              size_t ws_size, hipStream_t stream) {
  (void)in_sizes; (void)n_in; (void)out_size; (void)ws_size;
  const float* q = (const float*)d_in[0];
  const float* k = (const float*)d_in[1];
  const float* v = (const float*)d_in[2];
  const unsigned char* mask = (const unsigned char*)d_in[3];  // bool
  const int* edge = (const int*)d_in[4];
  const float* wq = (const float*)d_in[5], *bq = (const float*)d_in[6];
  const float* wk = (const float*)d_in[7], *bk = (const float*)d_in[8];
  const float* wv = (const float*)d_in[9], *bv = (const float*)d_in[10];
  const float* wo = (const float*)d_in[11], *bo = (const float*)d_in[12];
  const float* lng = (const float*)d_in[13], *lnb = (const float*)d_in[14];

  char* ws = (char*)d_ws;
  bf16_t* xq  = (bf16_t*)(ws + ((size_t)0 << 20));
  bf16_t* xk  = (bf16_t*)(ws + ((size_t)16 << 20));
  bf16_t* xv  = (bf16_t*)(ws + ((size_t)32 << 20));
  bf16_t* wqb = (bf16_t*)(ws + ((size_t)48 << 20));
  bf16_t* wkb = (bf16_t*)(ws + ((size_t)50 << 20));
  bf16_t* wvb = (bf16_t*)(ws + ((size_t)52 << 20));
  bf16_t* wob = (bf16_t*)(ws + ((size_t)54 << 20));
  bf16_t* qh  = (bf16_t*)(ws + ((size_t)56 << 20));
  bf16_t* kh  = (bf16_t*)(ws + ((size_t)72 << 20));
  bf16_t* vht = (bf16_t*)(ws + ((size_t)88 << 20));
  unsigned int* mbits = (unsigned int*)(ws + ((size_t)104 << 20));
  // Phase-based reuse (xq/xk/xv are dead after the projections):
  float*  xbuf = (float*)(ws + ((size_t)0 << 20));   // [M,1024] f32
  bf16_t* att  = (bf16_t*)(ws + ((size_t)32 << 20)); // [M,1024] bf16

  float* y_out = (float*)d_out;                    // [B,L,D]
  float* attn_out = y_out + (size_t)Bc * Lc * Dc;  // [H*B,L,L]

  // 0) bulk f32 -> bf16 conversion (activations + weights)
  cvt_bf16_kernel<<<(Mc * Dc) / 1024, 256, 0, stream>>>(q, xq);
  cvt_bf16_kernel<<<(Mc * Dc) / 1024, 256, 0, stream>>>(k, xk);
  cvt_bf16_kernel<<<(Mc * Dc) / 1024, 256, 0, stream>>>(v, xv);
  cvt_bf16_kernel<<<(Dc * Dc) / 1024, 256, 0, stream>>>(wq, wqb);
  cvt_bf16_kernel<<<(Dc * Dc) / 1024, 256, 0, stream>>>(wk, wkb);
  cvt_bf16_kernel<<<(Dc * Dc) / 1024, 256, 0, stream>>>(wv, wvb);
  cvt_bf16_kernel<<<(Dc * Dc) / 1024, 256, 0, stream>>>(wo, wob);

  // 1) QKV projections (TDM-staged weight tiles)
  const dim3 gproj(Dc / 64, Mc / 128);  // (16, 64)
  gemm_kernel<0><<<gproj, 256, 0, stream>>>(xq, wqb, bq, nullptr, qh, nullptr);
  gemm_kernel<0><<<gproj, 256, 0, stream>>>(xk, wkb, bk, nullptr, kh, nullptr);
  gemm_kernel<1><<<gproj, 256, 0, stream>>>(xv, wvb, bv, nullptr, vht, nullptr);

  // 2) mask pre-pack
  maskpack_kernel<<<(Bc * Lc * Lc / 32) / 256, 256, 0, stream>>>(edge, mask,
                                                                 mbits);

  // 3) fused attention (scores + softmax + attn store + P*V)
  const size_t attn_lds = 16 * Lc * 4 + 16 * Lc * 2;  // 192 KiB
  (void)hipFuncSetAttribute((const void*)attn_kernel,
                            hipFuncAttributeMaxDynamicSharedMemorySize,
                            (int)attn_lds);
  attn_kernel<<<Bc * Hc * (Lc / 16), 256, attn_lds, stream>>>(
      qh, kh, vht, mbits, attn_out, att);

  // 4) output projection + bias + residual, then LayerNorm
  gemm_kernel<2><<<gproj, 256, 0, stream>>>(att, wob, bo, q, nullptr, xbuf);
  ln_kernel<<<Mc, 256, 0, stream>>>(xbuf, lng, lnb, y_out);
}
